// GCN_798863917396
// MI455X (gfx1250) — compile-verified, matrix operand order checked by reading, not run on hardware
//
#include <hip/hip_runtime.h>
#include <hip/hip_bf16.h>
#include <cmath>

// Problem constants (match reference)
#define H_DIM   256
#define F_IN    512
#define NCLS    121
#define NLAYERS 4

typedef unsigned short u16;
typedef __attribute__((ext_vector_type(16))) __bf16 v16bf;
typedef __attribute__((ext_vector_type(8)))  float  v8f;

// float -> bf16 bits, round-to-nearest-even
__device__ __forceinline__ u16 bfbits(float f) {
    union { float f; unsigned u; } v; v.f = f;
    unsigned r = v.u + 0x7FFFu + ((v.u >> 16) & 1u);
    return (u16)(r >> 16);
}
__device__ __forceinline__ unsigned pack2bf(float a, float b) {
    return (unsigned)bfbits(a) | ((unsigned)bfbits(b) << 16);
}

// K index inside a 16x32 bf16 fragment held by VGPR-pair v for this lane
// (ISA 7.12.2): lanes 0-15: V0..3 -> K 0..7, V4..7 -> K 16..23 ; lanes 16-31: +8
__device__ __forceinline__ int frag_k(int v, int lane) {
    int khalf = (lane & 16) ? 8 : 0;
    return (v < 4) ? (khalf + 2 * v) : (16 + khalf + 2 * (v - 4));
}

// ---------------------------------------------------------------------------
// Weight pre-swizzle: f32 (K x ncolsReal, row-major) -> bf16 B-fragments.
// Fragment grid: (K/32) x nColTiles16; each fragment = 32 lanes x 16 bf16 (1KB).
// Zero-pads columns >= ncolsReal.
// ---------------------------------------------------------------------------
__global__ void pack_b_frags(const float* __restrict__ B, u16* __restrict__ dst,
                             int K, int ncolsReal, int nColTiles16) {
    int tid = blockIdx.x * blockDim.x + threadIdx.x;
    int lane = tid & 31;
    int frag = tid >> 5;
    int kTiles = K >> 5;
    if (frag >= kTiles * nColTiles16) return;
    int kt = frag / nColTiles16;
    int ct = frag % nColTiles16;
    int n = (ct << 4) + (lane & 15);
    bool ok = (n < ncolsReal);

    union { uint4 q[2]; u16 s[16]; } w;
#pragma unroll
    for (int v = 0; v < 8; ++v) {
        int k = (kt << 5) + frag_k(v, lane);
        float f0 = ok ? B[(size_t)k * ncolsReal + n] : 0.f;
        float f1 = ok ? B[(size_t)(k + 1) * ncolsReal + n] : 0.f;
        w.s[2 * v]     = bfbits(f0);
        w.s[2 * v + 1] = bfbits(f1);
    }
    uint4* dq = (uint4*)(dst + ((size_t)frag * 32 + lane) * 16);
    dq[0] = w.q[0];
    dq[1] = w.q[1];
}

// f32 -> bf16 row-major conversion (4 elements/thread)
__global__ void cvt_f32_bf16(const float* __restrict__ src, u16* __restrict__ dst, int n4) {
    int i = blockIdx.x * blockDim.x + threadIdx.x;
    if (i >= n4) return;
    float4 v = ((const float4*)src)[i];
    uint2 p;
    p.x = pack2bf(v.x, v.y);
    p.y = pack2bf(v.z, v.w);
    ((uint2*)dst)[i] = p;
}

__global__ void zero_f4(float* __restrict__ p, int n4) {
    int i = blockIdx.x * blockDim.x + threadIdx.x;
    if (i < n4) ((float4*)p)[i] = make_float4(0.f, 0.f, 0.f, 0.f);
}

// h[dst] += w * x[src] : one thread per (edge, 4-feature chunk)
__global__ void scatter_add_edges(const float* __restrict__ x,
                                  const int* __restrict__ src,
                                  const int* __restrict__ dst,
                                  const float* __restrict__ w,
                                  float* __restrict__ h, int E) {
    long long tid = (long long)blockIdx.x * blockDim.x + threadIdx.x;
    int chunk = (int)(tid & 63);
    long long e = tid >> 6;
    if (e >= E) return;
    int s = src[e], d = dst[e];
    float wt = w[e];
    float4 xv = *(const float4*)(x + (size_t)s * H_DIM + chunk * 4);
    float* hp = h + (size_t)d * H_DIM + chunk * 4;
    atomicAdd(hp + 0, wt * xv.x);
    atomicAdd(hp + 1, wt * xv.y);
    atomicAdd(hp + 2, wt * xv.z);
    atomicAdd(hp + 3, wt * xv.w);
}

// g = 0.5*h + 0.5*x0 (alpha=0.5), emit f32 + bf16; reset h to 0
__global__ void combine_reset(const float* __restrict__ h, const float* __restrict__ x0,
                              float* __restrict__ g, u16* __restrict__ gbf,
                              float* hout, int n4) {
    int i = blockIdx.x * blockDim.x + threadIdx.x;
    if (i >= n4) return;
    float4 hv = ((const float4*)h)[i];
    float4 xv = ((const float4*)x0)[i];
    float4 gv;
    gv.x = 0.5f * (hv.x + xv.x);
    gv.y = 0.5f * (hv.y + xv.y);
    gv.z = 0.5f * (hv.z + xv.z);
    gv.w = 0.5f * (hv.w + xv.w);
    ((float4*)g)[i] = gv;
    uint2 p;
    p.x = pack2bf(gv.x, gv.y);
    p.y = pack2bf(gv.z, gv.w);
    ((uint2*)gbf)[i] = p;
    ((float4*)hout)[i] = make_float4(0.f, 0.f, 0.f, 0.f);
}

// ---------------------------------------------------------------------------
// WMMA GEMM with async B staging through LDS.
// Block = 8 waves; each wave computes 32(M) x 64(N); block covers 256 rows of
// one 64-column strip. The strip's B panel (kTiles x 4KB) is copied to LDS
// once per block with GLOBAL_LOAD_ASYNC_TO_LDS_B128 (ASYNCcnt), then the
// K-loop reads fragments from LDS (ds_load_b128) and issues 8 WMMAs per step.
//  MODE 0 (lin0): out = relu(acc + bias)
//  MODE 1 (conv): out = relu((1-beta)*g + beta*acc + xold), also bf16 copy
//  MODE 2 (lin1): out = acc + bias (cols guarded to ncolsOut)
// ---------------------------------------------------------------------------
template<int MODE>
__device__ __forceinline__ void store_tile(v8f acc, int rbase, int mbase, int c,
                                           const float* __restrict__ bias,
                                           const float* __restrict__ gf32,
                                           const float* __restrict__ xold,
                                           float* __restrict__ out, int ldout,
                                           int ncolsOut, u16* __restrict__ outbf,
                                           int nRows, float beta) {
    if (c >= ncolsOut) return;
    float bval = (MODE != 1) ? bias[c] : 0.f;
#pragma unroll
    for (int v = 0; v < 8; ++v) {
        int rr = rbase + mbase + v;
        if (rr >= nRows) continue;
        float val = acc[v];
        size_t idx = (size_t)rr * ldout + c;
        if (MODE == 0) {
            val = fmaxf(val + bval, 0.f);
        } else if (MODE == 1) {
            val = (1.f - beta) * gf32[idx] + beta * val + xold[idx];
            val = fmaxf(val, 0.f);
        } else {
            val += bval;
        }
        out[idx] = val;
        if (MODE == 1) outbf[idx] = bfbits(val);
    }
}

template<int MODE>
__global__ __launch_bounds__(256) void gemm_wmma(
        const u16* __restrict__ Abf, int K,
        const u16* __restrict__ Bfrag, int nColTiles16,
        const float* __restrict__ bias,
        const float* __restrict__ gf32,
        const float* __restrict__ xold,
        float* __restrict__ out, int ldout, int ncolsOut,
        u16* __restrict__ outbf, int nRows, float beta) {
    extern __shared__ u16 bsh[];                 // kTiles * 4KB
    const int kTiles = K >> 5;
    const int nStrips = nColTiles16 >> 2;
    const int strip = blockIdx.x % nStrips;
    const int rowBlock = blockIdx.x / nStrips;
    const int ct4 = strip << 2;
    const int col0 = ct4 << 4;
    const int tid = threadIdx.x;

    // ---- async-stage B panel into LDS: 16B per thread per K-tile ----
    {
        unsigned lbase = (unsigned)(uintptr_t)(void*)bsh;   // low 32 bits = LDS offset
        const char* gbase = (const char*)Bfrag;
        for (int kt = 0; kt < kTiles; ++kt) {
            unsigned loff = lbase + (unsigned)(kt * 4096 + tid * 16);
            unsigned long long ga = (unsigned long long)
                (gbase + ((size_t)kt * nColTiles16 + ct4) * 1024 + (size_t)tid * 16);
            asm volatile("global_load_async_to_lds_b128 %0, %1, off"
                         :: "v"(loff), "v"(ga) : "memory");
        }
        asm volatile("s_wait_asynccnt 0" ::: "memory");
        __syncthreads();
    }

    const int lane = tid & 31;
    const int wave = tid >> 5;
    const int khalf = (lane & 16) ? 8 : 0;
    const int row0 = rowBlock * 256 + wave * 32;   // this wave's 32-row strip

    int ra0 = row0 + (lane & 15);
    int ra1 = row0 + 16 + (lane & 15);
    if (ra0 >= nRows) ra0 = nRows - 1;             // clamp; rows guarded at store
    if (ra1 >= nRows) ra1 = nRows - 1;
    const u16* arow0 = Abf + (size_t)ra0 * K + khalf;
    const u16* arow1 = Abf + (size_t)ra1 * K + khalf;
    const u16* bl = bsh + lane * 16;               // this lane's slice of fragment 0

    v8f a00 = {}, a01 = {}, a02 = {}, a03 = {};
    v8f a10 = {}, a11 = {}, a12 = {}, a13 = {};
    for (int kt = 0; kt < kTiles; ++kt) {
        // A fragments: two contiguous 16B loads each (layout-exact, 16-bit A 16x32)
        union { v16bf v; uint4 q[2]; } au0, au1;
        const u16* ap0 = arow0 + (kt << 5);
        const u16* ap1 = arow1 + (kt << 5);
        au0.q[0] = *(const uint4*)(ap0);
        au0.q[1] = *(const uint4*)(ap0 + 16);
        au1.q[0] = *(const uint4*)(ap1);
        au1.q[1] = *(const uint4*)(ap1 + 16);
        // B fragments from LDS (512 u16 apart; 2 x ds_load_b128 each)
        const u16* bkt = bl + kt * 2048;
        union { v16bf v; uint4 q[2]; } b0, b1, b2, b3;
        b0.q[0] = *(const uint4*)(bkt);        b0.q[1] = *(const uint4*)(bkt + 8);
        b1.q[0] = *(const uint4*)(bkt + 512);  b1.q[1] = *(const uint4*)(bkt + 520);
        b2.q[0] = *(const uint4*)(bkt + 1024); b2.q[1] = *(const uint4*)(bkt + 1032);
        b3.q[0] = *(const uint4*)(bkt + 1536); b3.q[1] = *(const uint4*)(bkt + 1544);
        a00 = __builtin_amdgcn_wmma_f32_16x16x32_bf16(false, au0.v, false, b0.v, (short)0, a00, false, false);
        a01 = __builtin_amdgcn_wmma_f32_16x16x32_bf16(false, au0.v, false, b1.v, (short)0, a01, false, false);
        a02 = __builtin_amdgcn_wmma_f32_16x16x32_bf16(false, au0.v, false, b2.v, (short)0, a02, false, false);
        a03 = __builtin_amdgcn_wmma_f32_16x16x32_bf16(false, au0.v, false, b3.v, (short)0, a03, false, false);
        a10 = __builtin_amdgcn_wmma_f32_16x16x32_bf16(false, au1.v, false, b0.v, (short)0, a10, false, false);
        a11 = __builtin_amdgcn_wmma_f32_16x16x32_bf16(false, au1.v, false, b1.v, (short)0, a11, false, false);
        a12 = __builtin_amdgcn_wmma_f32_16x16x32_bf16(false, au1.v, false, b2.v, (short)0, a12, false, false);
        a13 = __builtin_amdgcn_wmma_f32_16x16x32_bf16(false, au1.v, false, b3.v, (short)0, a13, false, false);
    }

    // C/D layout: lane<16 -> M=v, N=lane ; lane>=16 -> M=8+v, N=lane-16
    const int n = lane & 15;
    const int mbase = (lane & 16) ? 8 : 0;
    store_tile<MODE>(a00, row0,      mbase, col0 + n,      bias, gf32, xold, out, ldout, ncolsOut, outbf, nRows, beta);
    store_tile<MODE>(a01, row0,      mbase, col0 + 16 + n, bias, gf32, xold, out, ldout, ncolsOut, outbf, nRows, beta);
    store_tile<MODE>(a02, row0,      mbase, col0 + 32 + n, bias, gf32, xold, out, ldout, ncolsOut, outbf, nRows, beta);
    store_tile<MODE>(a03, row0,      mbase, col0 + 48 + n, bias, gf32, xold, out, ldout, ncolsOut, outbf, nRows, beta);
    store_tile<MODE>(a10, row0 + 16, mbase, col0 + n,      bias, gf32, xold, out, ldout, ncolsOut, outbf, nRows, beta);
    store_tile<MODE>(a11, row0 + 16, mbase, col0 + 16 + n, bias, gf32, xold, out, ldout, ncolsOut, outbf, nRows, beta);
    store_tile<MODE>(a12, row0 + 16, mbase, col0 + 32 + n, bias, gf32, xold, out, ldout, ncolsOut, outbf, nRows, beta);
    store_tile<MODE>(a13, row0 + 16, mbase, col0 + 48 + n, bias, gf32, xold, out, ldout, ncolsOut, outbf, nRows, beta);
}

extern "C" void kernel_launch(void* const* d_in, const int* in_sizes, int n_in,
                              void* d_out, int out_size, void* d_ws, size_t ws_size,
                              hipStream_t stream) {
    const float* x_in    = (const float*)d_in[0];
    const int*   esrc    = (const int*)  d_in[1];
    const int*   edst    = (const int*)  d_in[2];
    const float* ew      = (const float*)d_in[3];
    const float* lin0_w  = (const float*)d_in[4];
    const float* lin0_b  = (const float*)d_in[5];
    const float* conv_ws = (const float*)d_in[6];
    const float* lin1_w  = (const float*)d_in[7];
    const float* lin1_b  = (const float*)d_in[8];

    const int N = in_sizes[0] / F_IN;
    const int E = in_sizes[1];
    const size_t NH = (size_t)N * H_DIM;

    // ---- workspace layout (f32 then bf16 regions; all offsets 32B aligned) ----
    float* x0 = (float*)d_ws;          // N x H
    float* xa = x0 + NH;
    float* xb = xa + NH;
    float* h  = xb + NH;
    float* g  = h  + NH;
    u16* gbf   = (u16*)(g + NH);                 // N x H   bf16
    u16* xinbf = gbf + NH;                       // N x 512 bf16
    u16* xbf   = xinbf + (size_t)N * F_IN;       // N x H   bf16
    u16* wf0   = xbf + NH;                       // lin0: 16 kt x 16 ct x 512
    u16* wfc   = wf0 + 16 * 16 * 512;            // conv: 4 x (8 x 16 x 512)
    u16* wf1   = wfc + (size_t)NLAYERS * 8 * 16 * 512;  // lin1: 8 x 8 x 512

    const int n4 = (int)(NH / 4);
    const int eltBlocks = (n4 + 255) / 256;
    const int rowBlocks = (N + 255) / 256;

    // init: zero h, convert x_in to bf16, pack all weights into fragment order
    zero_f4<<<eltBlocks, 256, 0, stream>>>(h, n4);
    {
        int xin4 = (int)(((size_t)N * F_IN) / 4);
        cvt_f32_bf16<<<(xin4 + 255) / 256, 256, 0, stream>>>(x_in, xinbf, xin4);
    }
    pack_b_frags<<<(16 * 16 * 32 + 255) / 256, 256, 0, stream>>>(lin0_w, wf0, F_IN, H_DIM, 16);
    for (int l = 0; l < NLAYERS; ++l)
        pack_b_frags<<<(8 * 16 * 32 + 255) / 256, 256, 0, stream>>>(
            conv_ws + (size_t)l * H_DIM * H_DIM, wfc + (size_t)l * 8 * 16 * 512, H_DIM, H_DIM, 16);
    pack_b_frags<<<(8 * 8 * 32 + 255) / 256, 256, 0, stream>>>(lin1_w, wf1, H_DIM, NCLS, 8);

    // lin0: x0 = relu(x @ W0 + b0)   (K=512 -> 64KB LDS B panel)
    gemm_wmma<0><<<rowBlocks * 4, 256, 16 * 4096, stream>>>(
        xinbf, F_IN, wf0, 16, lin0_b, nullptr, nullptr,
        x0, H_DIM, H_DIM, nullptr, N, 0.f);

    const float* xcur = x0;
    float* pingpong[2] = { xa, xb };
    long long sThreads = (long long)E * 64;
    int sBlocks = (int)((sThreads + 255) / 256);

    for (int l = 0; l < NLAYERS; ++l) {
        float beta = logf(1.0f / (float)(l + 1) + 1.0f);
        scatter_add_edges<<<sBlocks, 256, 0, stream>>>(xcur, esrc, edst, ew, h, E);
        combine_reset<<<eltBlocks, 256, 0, stream>>>(h, x0, g, gbf, h, n4);
        float* xnext = pingpong[l & 1];
        gemm_wmma<1><<<rowBlocks * 4, 256, 8 * 4096, stream>>>(
            gbf, H_DIM, wfc + (size_t)l * 8 * 16 * 512, 16, nullptr, g, xcur,
            xnext, H_DIM, H_DIM, xbf, N, beta);
        xcur = xnext;
    }

    // lin1: out = x @ W1 + b1  (cols padded 121 -> 128 in fragments)
    gemm_wmma<2><<<rowBlocks * 2, 256, 8 * 4096, stream>>>(
        xbf, H_DIM, wf1, 8, lin1_b, nullptr, nullptr,
        (float*)d_out, NCLS, NCLS, nullptr, N, 0.f);
}
